// MemoryEfficientAttention_2181843386824
// MI455X (gfx1250) — matrix operand order repaired and checked by reading, not compiled, not run on hardware
//
#include <hip/hip_runtime.h>
#include <hip/hip_bf16.h>

// ---------------------------------------------------------------------------
// MemoryEfficientAttention fused pipeline for gfx1250 (MI455X, wave32, WMMA,
// async global->LDS).
//
//   K0: zero GroupNorm raw accumulators
//   K1 (fused): per (bh, chunk):
//         qkv_slice = W_qkv_slice @ x_chunk  (bf16 WMMA, all-LDS)
//         qs = softmax_i(q), ks = softmax_d(k)
//         S  = qs @ ks^T (32x32, K=1024 WMMA);  out = S @ v (WMMA)
//         -> attn stored TRANSPOSED [b][n][128] bf16 (vectorized stores)
//   K2: y = W_out @ attn^T + b_out (bf16 WMMA; async B-tile staging)
//       + fused GroupNorm partial sums (LDS + global atomics)
//   K3: finalize mean/invstd;  K4: normalize + affine -> d_out
// ---------------------------------------------------------------------------

typedef __attribute__((ext_vector_type(8)))  __bf16 v8bf;
typedef __attribute__((ext_vector_type(16))) __bf16 v16bf;
typedef __attribute__((ext_vector_type(8)))  float  v8f;

#define NQ    9216     // h*w
#define BATCH 8
#define CDIM  64
#define INNER 128
#define DH    32       // dim_head
#define NCH   9        // chunks
#define CH    1024     // chunk size

// ---- CDNA5 async global->LDS (ASYNCcnt) -----------------------------------
__device__ __forceinline__ void async_g2l_b128(void* lds, const void* g) {
  const unsigned loff = (unsigned)(size_t)lds;          // AS3 offset (low 32b)
  const unsigned long long ga = (unsigned long long)g;
  asm volatile("global_load_async_to_lds_b128 %0, %1, off"
               :: "v"(loff), "v"(ga) : "memory");
}
__device__ __forceinline__ void wait_async0() {
  asm volatile("s_wait_asynccnt 0x0" ::: "memory");
}

// A-matrix 16x32 bf16 fragment: lane holds row M=lane&15; element e maps to
// K = e + (e>=8 ? 8 : 0) + 8*hi  ->  two contiguous 8-half segments.
__device__ inline v16bf load_A_frag(const __bf16* row, int hi, int base_k) {
  const v8bf lo = *(const v8bf*)(row + base_k + hi * 8);
  const v8bf hh = *(const v8bf*)(row + base_k + hi * 8 + 16);
  v16bf a;
#pragma unroll
  for (int e = 0; e < 8; ++e) { a[e] = lo[e]; a[e + 8] = hh[e]; }
  return a;
}

// B-matrix 32x16 bf16 fragment: lane holds column N=lane&15; element e maps to
// K = e + (e>=8 ? 8 : 0) + 8*hi  ->  two contiguous 8-half segments.
__device__ inline v16bf load_B_frag(const __bf16* row, int hi, int base_k) {
  const v8bf lo = *(const v8bf*)(row + base_k + hi * 8);
  const v8bf hh = *(const v8bf*)(row + base_k + hi * 8 + 16);
  v16bf b;
#pragma unroll
  for (int e = 0; e < 8; ++e) { b[e] = lo[e]; b[e + 8] = hh[e]; }
  return b;
}

__device__ inline v8f wmma_bf16(v16bf a, v16bf b, v8f c) {
  return __builtin_amdgcn_wmma_f32_16x16x32_bf16(false, a, false, b,
                                                 (short)0, c, false, false);
}

// ---------------------------------------------------------------------------
// K1: fully fused QKV-projection + chunk attention.
// Grid (NCH, 32), 1024 threads = 32 waves, ~317 KB dynamic LDS.
//
// LDS map (halves unless noted):
//   [0, 73728)         Xt [1024][72]  (phase A/B)
//       aliased after Xt death: vcT [1024][40] @0, Sm [32][40] @40960
//   [73728, 80640)     Wl [96][72]   rows: 0-31 q, 32-63 k, 64-95 v
//   [80640, 113408)    qb [32][1024]   (aliased by oT [1024][40] after D)
//   [113408, 146176)   kb [32][1024]
//   [146176h ...)      Wtmp fp32 [96][64]  async staging (24576 B)
// ---------------------------------------------------------------------------
__global__ __launch_bounds__(1024) void fused_qkv_attn_kernel(
    const float* __restrict__ x, const float* __restrict__ Wqkv,
    __bf16* __restrict__ attn) {
  extern __shared__ char smem[];
  __bf16* Xt  = (__bf16*)smem;           // [1024][72]
  __bf16* vcT = Xt;                      // [1024][40] (aliases dead Xt)
  __bf16* Sm  = Xt + 1024 * 40;          // [32][40]   (aliases dead Xt)
  __bf16* Wl  = Xt + 1024 * 72;          // [96][72]
  __bf16* qb  = Wl + 96 * 72;            // [32][1024]
  __bf16* kb  = qb + 32 * 1024;          // [32][1024]
  __bf16* oT  = qb;                      // [1024][40] (aliases dead qb/kb)
  float*  Wtmp = (float*)(kb + 32 * 1024); // [96][64] fp32 async staging

  const int t  = threadIdx.x;
  const int ch = blockIdx.x;             // 0..8
  const int bh = blockIdx.y;             // 0..31
  const int b = bh >> 2, hd = bh & 3;
  const size_t nbase = (size_t)ch * CH;

  // ---- Phase A0: async-stage W slice (96 rows x 256 B) into Wtmp fp32 ----
  {
    const int f = t;                     // 1536 transfers of 16 B
    const int lr = f >> 4, seg = f & 15;
    if (f < 1536) {
      const int grow = (lr >> 5) * INNER + hd * DH + (lr & 31);
      async_g2l_b128(Wtmp + lr * 64 + seg * 4, Wqkv + grow * 64 + seg * 4);
    }
    const int f2 = t + 1024;
    const int lr2 = f2 >> 4, seg2 = f2 & 15;
    if (f2 < 1536) {
      const int grow = (lr2 >> 5) * INNER + hd * DH + (lr2 & 31);
      async_g2l_b128(Wtmp + lr2 * 64 + seg2 * 4, Wqkv + grow * 64 + seg2 * 4);
    }
  }
  // ---- Phase A1 (overlapped): stage x chunk transposed Xt[i][c], bf16 ----
  const float* xg = x + (size_t)b * CDIM * NQ + nbase;
#pragma unroll 4
  for (int c = 0; c < 64; ++c) Xt[t * 72 + c] = (__bf16)xg[(size_t)c * NQ + t];
  wait_async0();
  __syncthreads();
  // ---- Phase A2: convert W staging -> bf16 ----
  for (int idx = t; idx < 96 * 64; idx += 1024)
    Wl[(idx >> 6) * 72 + (idx & 63)] = (__bf16)Wtmp[idx];
  __syncthreads();

  const int wave = t >> 5, lane = t & 31, l15 = lane & 15, hi = lane >> 4;

  // ---- Phase B: qkv GEMM: 6 m-tiles (q0 q1 k0 k1 v0 v1) x 64 n-tiles ----
  v8f vacc[2][2];
#pragma unroll
  for (int q2 = 0; q2 < 2; ++q2) {
    const int nt = wave * 2 + q2;
    const __bf16* brow = Xt + (nt * 16 + l15) * 72;
    const v16bf b0 = load_B_frag(brow, hi, 0);
    const v16bf b1 = load_B_frag(brow, hi, 32);
#pragma unroll
    for (int mt = 0; mt < 6; ++mt) {
      const __bf16* arow = Wl + (mt * 16 + l15) * 72;
      v8f c = {};
      c = wmma_bf16(load_A_frag(arow, hi, 0), b0, c);
      c = wmma_bf16(load_A_frag(arow, hi, 32), b1, c);
      if (mt < 2) {            // q rows 0..31
#pragma unroll
        for (int r = 0; r < 8; ++r)
          qb[(mt * 16 + r + 8 * hi) * 1024 + nt * 16 + l15] = (__bf16)c[r];
      } else if (mt < 4) {     // k rows 0..31
#pragma unroll
        for (int r = 0; r < 8; ++r)
          kb[((mt - 2) * 16 + r + 8 * hi) * 1024 + nt * 16 + l15] = (__bf16)c[r];
      } else {                 // v: hold until Xt is dead
        vacc[q2][mt - 4] = c;
      }
    }
  }
  __syncthreads();             // all Xt reads done; qb/kb complete

  // ---- Phase C0: spill v into vcT[j][d] (aliases Xt) ----
#pragma unroll
  for (int q2 = 0; q2 < 2; ++q2) {
    const int nt = wave * 2 + q2;
#pragma unroll
    for (int mv = 0; mv < 2; ++mv) {
#pragma unroll
      for (int r = 0; r < 8; ++r)
        vcT[(nt * 16 + l15) * 40 + mv * 16 + r + 8 * hi] = (__bf16)vacc[q2][mv][r];
    }
  }
  // ---- Phase C1: ks column softmax over d (one thread per column) ----
  {
    float vals[32]; float mx = -3.4e38f;
#pragma unroll
    for (int d = 0; d < 32; ++d) { vals[d] = (float)kb[d * 1024 + t]; mx = fmaxf(mx, vals[d]); }
    float s = 0.f;
#pragma unroll
    for (int d = 0; d < 32; ++d) { vals[d] = __expf(vals[d] - mx); s += vals[d]; }
    const float inv = 1.f / s;
#pragma unroll
    for (int d = 0; d < 32; ++d) kb[d * 1024 + t] = (__bf16)(vals[d] * inv);
  }
  // ---- Phase C2: qs row softmax over chunk (one wave per row) ----
  {
    float vals[32]; float mx = -3.4e38f;
#pragma unroll
    for (int j = 0; j < 32; ++j) { vals[j] = (float)qb[wave * 1024 + j * 32 + lane]; mx = fmaxf(mx, vals[j]); }
#pragma unroll
    for (int m = 16; m >= 1; m >>= 1) mx = fmaxf(mx, __shfl_xor(mx, m, 32));
    float s = 0.f;
#pragma unroll
    for (int j = 0; j < 32; ++j) { vals[j] = __expf(vals[j] - mx); s += vals[j]; }
#pragma unroll
    for (int m = 16; m >= 1; m >>= 1) s += __shfl_xor(s, m, 32);
    const float inv = 1.f / s;
#pragma unroll
    for (int j = 0; j < 32; ++j) qb[wave * 1024 + j * 32 + lane] = (__bf16)(vals[j] * inv);
  }
  __syncthreads();

  // ---- Phase D: S = qs @ ks^T — 2x2 tiles, one wave each, K=1024 ----
  if (wave < 4) {
    const int mt = wave >> 1, nt = wave & 1;
    v8f c = {};
    for (int kk = 0; kk < 32; ++kk) {
      const v16bf a  = load_A_frag(qb + (mt * 16 + l15) * 1024, hi, kk * 32);
      const v16bf bb = load_B_frag(kb + (nt * 16 + l15) * 1024, hi, kk * 32);
      c = wmma_bf16(a, bb, c);
    }
#pragma unroll
    for (int r = 0; r < 8; ++r)
      Sm[(mt * 16 + r + 8 * hi) * 40 + nt * 16 + l15] = (__bf16)c[r];
  }
  __syncthreads();

  // ---- Phase E: out = S @ v — results into oT[j][d] (aliases dead qb) ----
  const v16bf a0 = load_A_frag(Sm + l15 * 40, hi, 0);
  const v16bf a1 = load_A_frag(Sm + (16 + l15) * 40, hi, 0);
#pragma unroll
  for (int q2 = 0; q2 < 2; ++q2) {
    const int jt = wave * 2 + q2;
    const v16bf bb = load_B_frag(vcT + (jt * 16 + l15) * 40, hi, 0);
    v8f c0 = {}, c1 = {};
    c0 = wmma_bf16(a0, bb, c0);
    c1 = wmma_bf16(a1, bb, c1);
#pragma unroll
    for (int r = 0; r < 8; ++r) {
      oT[(jt * 16 + l15) * 40 + r + 8 * hi]      = (__bf16)c0[r];
      oT[(jt * 16 + l15) * 40 + 16 + r + 8 * hi] = (__bf16)c1[r];
    }
  }
  __syncthreads();

  // ---- Phase F: vectorized transposed attn store: attn[b][n][128] ----
  __bf16* og = attn + ((size_t)b * NQ + nbase) * INNER + hd * DH;
#pragma unroll
  for (int it = 0; it < 4; ++it) {
    const int f = it * 1024 + t;
    const int j = f >> 2, seg = f & 3;
    const uint4 val = *(const uint4*)(oT + j * 40 + seg * 8);
    *(uint4*)(og + (size_t)j * INNER + seg * 8) = val;
  }
}

// ---------------------------------------------------------------------------
// K0: zero raw GroupNorm accumulators (graph-replay safe)
// ---------------------------------------------------------------------------
__global__ void zero_stats_kernel(float* __restrict__ stats_raw) {
  if (threadIdx.x < 128) stats_raw[threadIdx.x] = 0.f;
}

// ---------------------------------------------------------------------------
// K2: y = W_out @ attn^T + b_out  (M=64, K=128)  + fused GN partial sums.
// attn is [b][n][128] so the B-tile stages with NO transpose -> pure async
// global->LDS B128 copies (512 B fully-coalesced per instruction).
// Group of output element o is o/8 = mt*2 + hi (constant per fragment).
// ---------------------------------------------------------------------------
__global__ __launch_bounds__(256) void out_gemm_kernel(
    const __bf16* __restrict__ attn, const float* __restrict__ Wout,
    const float* __restrict__ bout, float* __restrict__ y,
    float* __restrict__ stats_raw) {
  extern __shared__ char smem[];
  __bf16* Wl = (__bf16*)smem;          // [64][136]
  __bf16* At = Wl + 64 * 136;          // [256][136]
  __shared__ float gs[16];             // [group][sum,sumsq]

  const int t  = threadIdx.x;
  const int b  = blockIdx.y;
  const int n0 = blockIdx.x * 256;

  // async-stage attn rows: 256 rows x 256 B, 16 B per lane transfer
  const __bf16* ab = attn + ((size_t)b * NQ + n0) * INNER;
  {
    const int seg = t & 15, r0 = t >> 4;
#pragma unroll
    for (int it = 0; it < 16; ++it) {
      const int i = it * 16 + r0;
      async_g2l_b128(At + i * 136 + seg * 8, ab + (size_t)i * INNER + seg * 8);
    }
  }
  if (t < 16) gs[t] = 0.f;
  for (int idx = t; idx < 64 * 128; idx += 256) {
    const int o = idx >> 7, c = idx & 127;
    Wl[o * 136 + c] = (__bf16)Wout[idx];
  }
  wait_async0();
  __syncthreads();

  const int wave = t >> 5, lane = t & 31, l15 = lane & 15, hi = lane >> 4;
  float* yb = y + (size_t)b * CDIM * NQ + n0;
  float a1[4] = {0.f, 0.f, 0.f, 0.f}, a2[4] = {0.f, 0.f, 0.f, 0.f};

#pragma unroll
  for (int q2 = 0; q2 < 2; ++q2) {
    const int nt = wave * 2 + q2;
    const __bf16* brow = At + (nt * 16 + l15) * 136;
    const v16bf b0 = load_B_frag(brow, hi, 0);
    const v16bf b1 = load_B_frag(brow, hi, 32);
    const v16bf b2 = load_B_frag(brow, hi, 64);
    const v16bf b3 = load_B_frag(brow, hi, 96);
#pragma unroll
    for (int mt = 0; mt < 4; ++mt) {
      const __bf16* arow = Wl + (mt * 16 + l15) * 136;
      v8f c = {};
      c = wmma_bf16(load_A_frag(arow, hi, 0),  b0, c);
      c = wmma_bf16(load_A_frag(arow, hi, 32), b1, c);
      c = wmma_bf16(load_A_frag(arow, hi, 64), b2, c);
      c = wmma_bf16(load_A_frag(arow, hi, 96), b3, c);
#pragma unroll
      for (int r = 0; r < 8; ++r) {
        const int o = mt * 16 + r + 8 * hi;
        const float v = c[r] + bout[o];
        yb[(size_t)o * NQ + nt * 16 + l15] = v;
        a1[mt] += v;
        a2[mt] += v * v;
      }
    }
  }
#pragma unroll
  for (int mt = 0; mt < 4; ++mt) {
    const int g = mt * 2 + hi;
    atomicAdd(&gs[g * 2],     a1[mt]);
    atomicAdd(&gs[g * 2 + 1], a2[mt]);
  }
  __syncthreads();
  if (t < 8) {
    atomicAdd(&stats_raw[(b * 8 + t) * 2],     gs[t * 2]);
    atomicAdd(&stats_raw[(b * 8 + t) * 2 + 1], gs[t * 2 + 1]);
  }
}

// ---------------------------------------------------------------------------
// K3: finalize mean/invstd per (b, group)
// ---------------------------------------------------------------------------
__global__ void gn_finalize_kernel(const float* __restrict__ stats_raw,
                                   float* __restrict__ stats) {
  const int i = threadIdx.x;
  if (i < 64) {
    const float invN = 1.f / (8.f * NQ);
    const float mu  = stats_raw[i * 2] * invN;
    const float var = stats_raw[i * 2 + 1] * invN - mu * mu;
    stats[i * 2]     = mu;
    stats[i * 2 + 1] = rsqrtf(var + 1e-5f);
  }
}

// ---------------------------------------------------------------------------
// K4: normalize + affine
// ---------------------------------------------------------------------------
__global__ __launch_bounds__(256) void gn_norm_kernel(
    const float* __restrict__ y, const float* __restrict__ stats,
    const float* __restrict__ gamma, const float* __restrict__ beta,
    float* __restrict__ out) {
  const size_t idx = (size_t)blockIdx.x * 256 + threadIdx.x;
  if (idx >= (size_t)BATCH * CDIM * NQ) return;
  const int c  = (int)((idx / NQ) & 63);
  const int bg = (int)(idx / ((size_t)8 * NQ));   // == b*8 + c/8
  const float mu = stats[bg * 2], is = stats[bg * 2 + 1];
  out[idx] = (y[idx] - mu) * is * gamma[c] + beta[c];
}

// ---------------------------------------------------------------------------
extern "C" void kernel_launch(void* const* d_in, const int* in_sizes, int n_in,
                              void* d_out, int out_size, void* d_ws, size_t ws_size,
                              hipStream_t stream) {
  (void)in_sizes; (void)n_in; (void)out_size; (void)ws_size;
  const float* x     = (const float*)d_in[0];
  const float* Wqkv  = (const float*)d_in[1];
  const float* Wout  = (const float*)d_in[2];
  const float* bout  = (const float*)d_in[3];
  const float* gamma = (const float*)d_in[4];
  const float* beta  = (const float*)d_in[5];
  float* out = (float*)d_out;

  // Workspace layout (~38 MB)
  const size_t ATTN_BYTES = (size_t)BATCH * NQ * INNER * sizeof(__bf16);  // 18.9 MB
  const size_t Y_BYTES    = (size_t)BATCH * CDIM * NQ * sizeof(float);    // 18.9 MB
  char* ws = (char*)d_ws;
  __bf16* attn      = (__bf16*)ws;                  // transposed [b][n][128]
  float*  y         = (float*)(ws + ATTN_BYTES);
  float*  stats_raw = (float*)(ws + ATTN_BYTES + Y_BYTES);
  float*  stats     = stats_raw + 128;

  const size_t lds1 = (size_t)(1024 * 72 + 96 * 72 + 32 * 1024 + 32 * 1024)
                      * sizeof(__bf16) + 96 * 64 * sizeof(float);         // 316928
  const size_t lds2 = (size_t)(64 * 136 + 256 * 136) * sizeof(__bf16);    // 87040

  zero_stats_kernel<<<1, 128, 0, stream>>>(stats_raw);
  fused_qkv_attn_kernel<<<dim3(NCH, BATCH * 4), 1024, lds1, stream>>>(x, Wqkv, attn);
  out_gemm_kernel<<<dim3(NQ / 256, BATCH), 256, lds2, stream>>>(attn, Wout, bout, y, stats_raw);
  gn_finalize_kernel<<<1, 64, 0, stream>>>(stats_raw, stats);
  gn_norm_kernel<<<(BATCH * CDIM * NQ) / 256, 256, 0, stream>>>(y, stats, gamma, beta, out);
}
